// FluvialAttention_30751965839621
// MI455X (gfx1250) — matrix operand order repaired and checked by reading, not compiled, not run on hardware
//
#include <hip/hip_runtime.h>
#include <hip/hip_bf16.h>

// ---------------------------------------------------------------------------
// CDNA5 (gfx1250) fused window-attention:
//   K0: weight pre-convert f32 -> f16 (qkv_w, proj_w) into workspace
//   K1: qkv = x @ qkv_w^T + qkv_b            (f32 in, f16 out)   WMMA f16
//   K2: per (window,head) attention          (f16 in, f16 out)   WMMA f16
//   K3: out = attn @ proj_w^T + proj_b       (f16 in, f32 out)   WMMA f16
// B tiles double-buffered in LDS; staged via GLOBAL_LOAD_ASYNC_TO_LDS_B128
// (ASYNCcnt-tracked) with sync v8h fallback.
// ---------------------------------------------------------------------------

typedef __attribute__((ext_vector_type(16))) _Float16 v16h;
typedef __attribute__((ext_vector_type(8)))  _Float16 v8h;
typedef __attribute__((ext_vector_type(4)))  _Float16 v4h;
typedef __attribute__((ext_vector_type(8)))  float    v8f;
typedef __attribute__((ext_vector_type(4)))  int      v4i;

#define WMMA_F16(a, b, c) \
  __builtin_amdgcn_wmma_f32_16x16x32_f16(false, (a), false, (b), (short)0, (c), false, false)

#if __has_builtin(__builtin_amdgcn_global_load_async_to_lds_b128)
#define HAVE_ASYNC_LDS 1
#else
#define HAVE_ASYNC_LDS 0
#endif

// async global->LDS copy of 16 bytes per lane (ASYNCcnt-tracked)
__device__ inline void g2l_b128(const void* g, void* l) {
#if HAVE_ASYNC_LDS
  __builtin_amdgcn_global_load_async_to_lds_b128(
      (__attribute__((address_space(1))) v4i*)(void*)g,
      (__attribute__((address_space(3))) v4i*)l, 0, 0);
#else
  *(v8h*)l = *(const v8h*)g;
#endif
}

__device__ inline void async_wait0() {
#if HAVE_ASYNC_LDS
#if __has_builtin(__builtin_amdgcn_s_wait_asynccnt)
  __builtin_amdgcn_s_wait_asynccnt(0);
#else
  asm volatile("s_wait_asynccnt 0" ::: "memory");
#endif
#endif
}

// A-matrix fragment (16x32 f16, M x K). ISA layout: lane L holds row M=L&15;
// halves 0..7  -> K = 8*(L>>4) + e,  halves 8..15 -> K = 16 + 8*(L>>4) + (e-8).
__device__ inline v16h frag_a(const _Float16* row, int h2) {
  const v8h lo = *(const v8h*)(row + 8 * h2);
  const v8h hi = *(const v8h*)(row + 16 + 8 * h2);
  return __builtin_shufflevector(lo, hi, 0, 1, 2, 3, 4, 5, 6, 7,
                                         8, 9, 10, 11, 12, 13, 14, 15);
}

// B-matrix fragment (32x16 f16, K x N). Lane L holds column N=L&15;
// halves 0..7 -> K = 8*(L>>4) + e, halves 8..15 -> K = 16 + 8*(L>>4) + (e-8).
__device__ inline v16h frag_b(const _Float16* col, int h2) {
  const v8h lo = *(const v8h*)(col + 8 * h2);
  const v8h hi = *(const v8h*)(col + 16 + 8 * h2);
  return __builtin_shufflevector(lo, hi, 0, 1, 2, 3, 4, 5, 6, 7,
                                         8, 9, 10, 11, 12, 13, 14, 15);
}

// ---------------------------------------------------------------------------
// K0: f32 -> f16 weight conversion (vectorized x4)
// ---------------------------------------------------------------------------
__global__ __launch_bounds__(256) void cvt_f32_f16_kernel(
    const float* __restrict__ src, _Float16* __restrict__ dst, int n4) {
  for (int i = blockIdx.x * 256 + threadIdx.x; i < n4; i += gridDim.x * 256) {
    const float4 v = ((const float4*)src)[i];
    v4h h = {(_Float16)v.x, (_Float16)v.y, (_Float16)v.z, (_Float16)v.w};
    ((v4h*)dst)[i] = h;
  }
}

// ---------------------------------------------------------------------------
// Generic GEMM: Out[M,N] = A[M,K=384] @ Wh[N,K]^T + bias (Wh already f16).
// Block = 128 threads (4 waves), BM=64, BN=64, BK=32 per WMMA step.
// A panel (64x384 f16) staged once; B 64x32 tiles double-buffered per k-step.
// ---------------------------------------------------------------------------
constexpr int GK = 384;      // reduction dim (both GEMMs)
constexpr int GKT = 12;      // 384 / 32
constexpr int PA = 392;      // A LDS pitch in halves (bank-conflict-free frags)
constexpr int PB = 40;       // B LDS pitch in halves

// stage one 64x32 f16 B tile: 256 x 16-byte chunks, 2 per thread
__device__ inline void stage_b(_Float16* __restrict__ dst,
                               const _Float16* __restrict__ Wh,
                               int nb, int kt, int tid) {
  for (int i = tid; i < 256; i += 128) {
    const int j = i >> 2, c8 = i & 3;
    const _Float16* src = Wh + (size_t)(nb * 64 + j) * GK + kt * 32 + c8 * 8;
    g2l_b128(src, dst + j * PB + c8 * 8);
    if (kt + 2 < GKT) __builtin_prefetch(src + 64, 0, 1);  // k-step after next
  }
}

template <bool AF16, bool OF16>
__global__ __launch_bounds__(128) void wmma_gemm_kernel(
    const void* __restrict__ Aptr, const _Float16* __restrict__ Wh,
    const float* __restrict__ bias, void* __restrict__ Outp, int N) {
  __shared__ alignas(16) _Float16 Ah[64 * PA];   // 50176 B
  __shared__ alignas(16) _Float16 Bh0[64 * PB];  //  5120 B
  __shared__ alignas(16) _Float16 Bh1[64 * PB];  //  5120 B

  const int tid = threadIdx.x;
  const int wave = tid >> 5;
  const int lane = tid & 31;
  const int h2 = lane >> 4;
  const int l16 = lane & 15;
  const long mbase = (long)blockIdx.x * 64;

  // ---- stage A panel: 64 rows x 384 ----
  if constexpr (AF16) {
    const _Float16* A = (const _Float16*)Aptr;
    for (int i = tid; i < 64 * 48; i += 128) {  // 48 x 16B per row
      const int r = i / 48, c8 = i % 48;
      g2l_b128(A + (mbase + r) * GK + c8 * 8, Ah + r * PA + c8 * 8);
    }
  } else {
    const float* A = (const float*)Aptr;
    for (int i = tid; i < 64 * 96; i += 128) {  // 96 x float4 per row
      const int r = i / 96, c4 = i % 96;
      const float4 v = *(const float4*)(A + (mbase + r) * GK + c4 * 4);
      v4h hv = {(_Float16)v.x, (_Float16)v.y, (_Float16)v.z, (_Float16)v.w};
      *(v4h*)(Ah + r * PA + c4 * 4) = hv;
    }
  }

  const int nblocks = N / 64;
  for (int nb = 0; nb < nblocks; ++nb) {
    v8f acc[4];
    const v8f vzero = {0.f, 0.f, 0.f, 0.f, 0.f, 0.f, 0.f, 0.f};
#pragma unroll
    for (int mf = 0; mf < 4; ++mf) acc[mf] = vzero;

    stage_b(Bh0, Wh, nb, 0, tid);   // prologue: first tile into buffer 0
    async_wait0();
    __syncthreads();                // A panel + B0 visible to all waves

    for (int kt = 0; kt < GKT; ++kt) {
      _Float16* cur = (kt & 1) ? Bh1 : Bh0;
      _Float16* nxt = (kt & 1) ? Bh0 : Bh1;
      if (kt + 1 < GKT) stage_b(nxt, Wh, nb, kt + 1, tid);  // overlap with WMMA

      const v16h bf = frag_b(cur + (wave * 16 + l16) * PB, h2);
#pragma unroll
      for (int mf = 0; mf < 4; ++mf) {
        const v16h af = frag_a(Ah + (mf * 16 + l16) * PA + kt * 32, h2);
        acc[mf] = WMMA_F16(af, bf, acc[mf]);
      }
      async_wait0();
      __syncthreads();  // next tile staged everywhere; cur free for reuse
    }

    // ---- epilogue: bias + store (C/D layout: lane = column, VGPR r = row) ----
    const int gcol = nb * 64 + wave * 16 + l16;
    const float bv = bias[gcol];
#pragma unroll
    for (int mf = 0; mf < 4; ++mf) {
#pragma unroll
      for (int r = 0; r < 8; ++r) {
        const long grow = mbase + mf * 16 + 8 * h2 + r;
        const float val = acc[mf][r] + bv;
        if constexpr (OF16)
          ((_Float16*)Outp)[grow * (long)N + gcol] = (_Float16)val;
        else
          ((float*)Outp)[grow * (long)N + gcol] = val;
      }
    }
  }
}

// ---------------------------------------------------------------------------
// K2: attention, one block (128 thr / 4 waves) per (window b, head h).
// N=49 tokens padded to 64, d=32. S = (Q*scale) K^T + bias + mask; softmax;
// O = P V. All matmuls via v_wmma_f32_16x16x32_f16.
// ---------------------------------------------------------------------------
constexpr int PQ = 40;  // Q/K LDS pitch (halves)
constexpr int PV = 72;  // Vt / P LDS pitch (halves)
constexpr int PS = 65;  // S LDS pitch (floats)

__global__ __launch_bounds__(128) void attn_kernel(
    const _Float16* __restrict__ qkv, const float* __restrict__ mask,
    const float* __restrict__ bias_table, const int* __restrict__ rel_index,
    _Float16* __restrict__ attn_out) {
  __shared__ alignas(16) _Float16 Qh[64 * PQ];
  __shared__ alignas(16) _Float16 Kh[64 * PQ];
  __shared__ alignas(16) _Float16 Vt[32 * PV];  // transposed: [dim][token]
  __shared__ alignas(16) _Float16 Ph[64 * PV];  // softmax(P), f16
  __shared__ float Sf[64 * PS];

  const int tid = threadIdx.x;
  const int wave = tid >> 5;
  const int lane = tid & 31;
  const int h2 = lane >> 4;
  const int l16 = lane & 15;
  const int b = blockIdx.x / 12;
  const int h = blockIdx.x % 12;
  const float scale = 0.17677669529663687f;  // 32^-0.5
  const v8f vzero = {0.f, 0.f, 0.f, 0.f, 0.f, 0.f, 0.f, 0.f};

  // zero Q/K/Vt so padded rows/cols (49..63) can't feed NaNs into WMMA
  for (int i = tid; i < 64 * PQ; i += 128) { Qh[i] = (_Float16)0.f; Kh[i] = (_Float16)0.f; }
  for (int i = tid; i < 32 * PV; i += 128) Vt[i] = (_Float16)0.f;
  __syncthreads();

  // stage q (pre-scaled), k, v^T from the fused qkv buffer (row-major, ld=1152)
  const _Float16* base = qkv + (size_t)b * 49 * 1152 + (size_t)h * 32;
  for (int i = tid; i < 49 * 32; i += 128) {
    const int t = i >> 5, dd = i & 31;
    const size_t row = (size_t)t * 1152;
    Qh[t * PQ + dd] = (_Float16)((float)base[row + dd] * scale);
    Kh[t * PQ + dd] = base[row + 384 + dd];
    Vt[dd * PV + t] = base[row + 768 + dd];
  }
  __syncthreads();

  // ---- S = Q K^T : 64x64 in 4x4 tiles; wave w owns column tile w; K=32 -> 1 wmma
  {
    const v16h bf = frag_b(Kh + (wave * 16 + l16) * PQ, h2);
#pragma unroll
    for (int mf = 0; mf < 4; ++mf) {
      const v16h af = frag_a(Qh + (mf * 16 + l16) * PQ, h2);
      const v8f s = WMMA_F16(af, bf, vzero);
#pragma unroll
      for (int r = 0; r < 8; ++r)
        Sf[(mf * 16 + 8 * h2 + r) * PS + wave * 16 + l16] = s[r];
    }
  }
  __syncthreads();

  // ---- softmax per row (rel-bias gather + mask), write f16 P ----
  if (tid < 64) {
    const int r = tid;
    if (r < 49) {
      const int* ridx = rel_index + r * 49;
      const float* mrow = mask + ((size_t)b * 49 + r) * 49;
      float mx = -1e30f;
      for (int c = 0; c < 49; ++c) {
        const float v = Sf[r * PS + c] + bias_table[ridx[c] * 12 + h] + mrow[c];
        Sf[r * PS + c] = v;
        mx = fmaxf(mx, v);
      }
      float sum = 0.f;
      for (int c = 0; c < 49; ++c) {
        const float e = __expf(Sf[r * PS + c] - mx);
        Sf[r * PS + c] = e;
        sum += e;
      }
      const float inv = 1.0f / sum;
      for (int c = 0; c < 49; ++c) Ph[r * PV + c] = (_Float16)(Sf[r * PS + c] * inv);
      for (int c = 49; c < 64; ++c) Ph[r * PV + c] = (_Float16)0.f;
    } else {
      for (int c = 0; c < 64; ++c) Ph[r * PV + c] = (_Float16)0.f;
    }
  }
  __syncthreads();

  // ---- O = P V : 64x32; wave w owns row tile w; K=64 -> 2 steps; 2 N-tiles
  v8f o0 = vzero, o1 = vzero;
#pragma unroll
  for (int kt = 0; kt < 2; ++kt) {
    const v16h af = frag_a(Ph + (wave * 16 + l16) * PV + kt * 32, h2);
    const v16h b0 = frag_b(Vt + (0 + l16) * PV + kt * 32, h2);
    const v16h b1 = frag_b(Vt + (16 + l16) * PV + kt * 32, h2);
    o0 = WMMA_F16(af, b0, o0);
    o1 = WMMA_F16(af, b1, o1);
  }
#pragma unroll
  for (int r = 0; r < 8; ++r) {
    const int token = wave * 16 + 8 * h2 + r;
    if (token < 49) {
      const size_t orow = ((size_t)b * 49 + token) * 384 + (size_t)h * 32;
      attn_out[orow + l16] = (_Float16)o0[r];
      attn_out[orow + 16 + l16] = (_Float16)o1[r];
    }
  }
}

// ---------------------------------------------------------------------------
extern "C" void kernel_launch(void* const* d_in, const int* in_sizes, int n_in,
                              void* d_out, int out_size, void* d_ws, size_t ws_size,
                              hipStream_t stream) {
  (void)in_sizes; (void)n_in; (void)out_size; (void)ws_size;
  const float* x          = (const float*)d_in[0];  // (4096, 49, 384)
  const float* mask       = (const float*)d_in[1];  // (4096, 49, 49)
  const float* qkv_w      = (const float*)d_in[2];  // (1152, 384)
  const float* qkv_b      = (const float*)d_in[3];  // (1152,)
  const float* proj_w     = (const float*)d_in[4];  // (384, 384)
  const float* proj_b     = (const float*)d_in[5];  // (384,)
  const float* bias_table = (const float*)d_in[6];  // (169, 12)
  const int*   rel_index  = (const int*)d_in[7];    // (49, 49)

  const long M = 200704;  // 4096 * 49
  _Float16* qkv  = (_Float16*)d_ws;                     // M x 1152 f16
  _Float16* attn = qkv  + (size_t)M * 1152;             // M x 384  f16
  _Float16* wq   = attn + (size_t)M * 384;              // 1152 x 384 f16
  _Float16* wp   = wq   + (size_t)1152 * 384;           // 384 x 384 f16

  // K0: convert weights to f16 once (instead of per-workgroup cvt)
  cvt_f32_f16_kernel<<<216, 256, 0, stream>>>(qkv_w, wq, 1152 * 384 / 4);
  cvt_f32_f16_kernel<<<72, 256, 0, stream>>>(proj_w, wp, 384 * 384 / 4);

  // K1: fused qkv projection (f32 -> f16 out)
  wmma_gemm_kernel<false, true><<<M / 64, 128, 0, stream>>>(
      (const void*)x, wq, qkv_b, (void*)qkv, 1152);

  // K2: windowed attention per (window, head)
  attn_kernel<<<4096 * 12, 128, 0, stream>>>(qkv, mask, bias_table, rel_index, attn);

  // K3: output projection (f16 -> f32 out)
  wmma_gemm_kernel<true, false><<<M / 64, 128, 0, stream>>>(
      (const void*)attn, wp, proj_b, d_out, 384);
}
